// MultiHeadAttention_6442450944647
// MI455X (gfx1250) — compile-verified, matrix operand order checked by reading, not run on hardware
//
#include <hip/hip_runtime.h>
#include <hip/hip_bf16.h>
#include <stdint.h>

// ---------------------------------------------------------------------------
// Types for gfx1250 WMMA (wave32): 16x16x32 f16 -> f32
// ---------------------------------------------------------------------------
typedef __attribute__((ext_vector_type(16))) _Float16 v16h;
typedef __attribute__((ext_vector_type(8)))  _Float16 v8h;
typedef __attribute__((ext_vector_type(8)))  float    v8f;

#define HIDDEN   2048
#define NHEADS   16
#define HDIM     128
#define BATCH    2
#define SEQ      2048
#define MROWS    (BATCH * SEQ)   // 4096 rows for all GEMMs

__device__ __forceinline__ v8f wmma_f16(v16h a, v16h b, v8f c) {
  // D = A(16x32 f16) * B(32x16 f16) + C(16x16 f32)
  return __builtin_amdgcn_wmma_f32_16x16x32_f16(
      /*neg_a=*/false, a, /*neg_b=*/false, b,
      /*c_mod=*/(short)0, c, /*reuse_a=*/false, /*reuse_b=*/false);
}

// Load a 16x32 (rows x K) f16 fragment from row-major memory.
// Per ISA 16-bit A/B layout: lane L holds row (L&15); per-lane K coverage is
//   K = kb + {0..7} and kb + 16 + {0..7}  with kb = 8*(L>>4)
// -> two contiguous 16-byte loads per lane (global_load_b128).
__device__ __forceinline__ v16h load_frag(const _Float16* __restrict__ base,
                                          int ld, int k0, int lane) {
  const _Float16* p = base + (size_t)(lane & 15) * (size_t)ld
                           + (size_t)(k0 + ((lane >> 4) << 3));
  v8h lo = *(const v8h*)p;
  v8h hi = *(const v8h*)(p + 16);
  v16h r;
#pragma unroll
  for (int i = 0; i < 8; ++i) { r[i] = lo[i]; r[i + 8] = hi[i]; }
  return r;
}

// ---------------------------------------------------------------------------
// Elementwise: fp32 -> f16 cast
// ---------------------------------------------------------------------------
__global__ void cast_f32_to_f16(const float* __restrict__ in,
                                _Float16* __restrict__ out, int n) {
  int i = blockIdx.x * 256 + threadIdx.x;
  if (i < n) out[i] = (_Float16)in[i];
}

// Tiled transpose + cast: Wt[n][k] = (f16)W[k][n], square dim = HIDDEN
__global__ void transpose_cast_w(const float* __restrict__ W,
                                 _Float16* __restrict__ Wt) {
  __shared__ float tile[32][33];
  const int n = HIDDEN;
  int bx = blockIdx.x * 32, by = blockIdx.y * 32;
  int tx = threadIdx.x, ty = threadIdx.y;  // 32 x 8
#pragma unroll
  for (int i = 0; i < 32; i += 8)
    tile[ty + i][tx] = W[(size_t)(by + ty + i) * n + bx + tx];
  __syncthreads();
#pragma unroll
  for (int i = 0; i < 32; i += 8)
    Wt[(size_t)(bx + ty + i) * n + by + tx] = (_Float16)tile[tx][ty + i];
}

// V (B,S,H*D) f16 -> Vt (B,H,D,S) f16 so PV B-fragments are contiguous
__global__ void transpose_v(const _Float16* __restrict__ V,
                            _Float16* __restrict__ Vt) {
  size_t i = (size_t)blockIdx.x * 256 + threadIdx.x;  // over B*S*HIDDEN
  if (i >= (size_t)BATCH * SEQ * HIDDEN) return;
  int c = (int)(i % HIDDEN);
  size_t bs = i / HIDDEN;
  int s = (int)(bs % SEQ);
  int b = (int)(bs / SEQ);
  int h = c >> 7, d = c & 127;
  Vt[(((size_t)(b * NHEADS + h) * HDIM) + d) * SEQ + s] = V[i];
}

// ---------------------------------------------------------------------------
// WMMA GEMM: C(MxN) = A(MxK f16, row-major) * Bt(NxK f16, row-major)^T + bias
// Block = 256 threads = 8 waves (2 x 4). Wave tile = 32x64 (2x4 WMMA accs):
// 6 fragment loads (12 x b128) per 8 WMMAs -> 1.5 loads/WMMA, WMMA-pipe bound.
// Block tile = 64(M) x 256(N). Weights (8MB f16) live in the 192MB L2; the
// K-loop prefetches 256B ahead (global_prefetch_b8).
// ---------------------------------------------------------------------------
template <typename OutT>
__global__ __launch_bounds__(256)
void gemm_bias_kernel(const _Float16* __restrict__ A,
                      const _Float16* __restrict__ Bt,
                      const float* __restrict__ bias,
                      OutT* __restrict__ C, int M, int N, int Kd) {
  const int lane = threadIdx.x & 31;
  const int w    = threadIdx.x >> 5;
  const int wm   = w & 1;       // 2 waves along M
  const int wn   = w >> 1;      // 4 waves along N
  const int row0 = blockIdx.y * 64  + wm * 32;
  const int col0 = blockIdx.x * 256 + wn * 64;

  v8f acc[2][4] = {};
  const _Float16* Ar[2];
  const _Float16* Bc[4];
#pragma unroll
  for (int i = 0; i < 2; ++i) Ar[i] = A  + (size_t)(row0 + i * 16) * Kd;
#pragma unroll
  for (int j = 0; j < 4; ++j) Bc[j] = Bt + (size_t)(col0 + j * 16) * Kd;

#pragma unroll 2
  for (int k = 0; k < Kd; k += 32) {
    // prefetch next 4 K-steps of each stream (global_prefetch_b8)
    __builtin_prefetch(Ar[0] + (size_t)(lane & 15) * Kd + k + 128, 0, 1);
    __builtin_prefetch(Bc[0] + (size_t)(lane & 15) * Kd + k + 128, 0, 1);

    v16h a0 = load_frag(Ar[0], Kd, k, lane);
    v16h a1 = load_frag(Ar[1], Kd, k, lane);
    v16h b0 = load_frag(Bc[0], Kd, k, lane);
    v16h b1 = load_frag(Bc[1], Kd, k, lane);
    v16h b2 = load_frag(Bc[2], Kd, k, lane);
    v16h b3 = load_frag(Bc[3], Kd, k, lane);
    acc[0][0] = wmma_f16(a0, b0, acc[0][0]);
    acc[0][1] = wmma_f16(a0, b1, acc[0][1]);
    acc[0][2] = wmma_f16(a0, b2, acc[0][2]);
    acc[0][3] = wmma_f16(a0, b3, acc[0][3]);
    acc[1][0] = wmma_f16(a1, b0, acc[1][0]);
    acc[1][1] = wmma_f16(a1, b1, acc[1][1]);
    acc[1][2] = wmma_f16(a1, b2, acc[1][2]);
    acc[1][3] = wmma_f16(a1, b3, acc[1][3]);
  }

  // C/D layout: VGPR r -> row r + 8*(lane>>4); col = lane&15
  const int nlo = lane & 15;
  const int mhi = (lane >> 4) << 3;
#pragma unroll
  for (int i = 0; i < 2; ++i)
#pragma unroll
    for (int j = 0; j < 4; ++j) {
      const int ncol = col0 + j * 16 + nlo;
      const float bb = bias[ncol];
#pragma unroll
      for (int r = 0; r < 8; ++r) {
        const int m = row0 + i * 16 + r + mhi;
        C[(size_t)m * N + ncol] = (OutT)(acc[i][j][r] + bb);
      }
    }
}

// ---------------------------------------------------------------------------
// Flash attention: block = 128 threads = 4 independent waves; each wave owns
// 16 query rows of one (b,h) and a private 1KB LDS slice (no block sync).
// Q frags cached in VGPRs (16x128). Loop over 64 key tiles of 32:
//   scores (2 x 16x16 via 8 WMMA), online softmax (shfl_xor over 16 lanes),
//   P relayout C->A via LDS tile, 8 PV WMMAs against Vt (D x S).
// ---------------------------------------------------------------------------
__global__ __launch_bounds__(128)
void attn_kernel(const _Float16* __restrict__ Q,
                 const _Float16* __restrict__ K,
                 const _Float16* __restrict__ Vt,
                 _Float16* __restrict__ O) {
  __shared__ _Float16 pl[4][16][32];  // per-wave P relayout buffers

  const int lane = threadIdx.x & 31;
  const int wv   = threadIdx.x >> 5;
  const int q0   = (blockIdx.x * 4 + wv) * 16;
  const int h    = blockIdx.y;
  const int b    = blockIdx.z;
  const float scale = 0.08838834764831845f;  // 1/sqrt(128)

  const _Float16* Qb  = Q  + (size_t)b * SEQ * HIDDEN + (size_t)q0 * HIDDEN + h * HDIM;
  const _Float16* Kb  = K  + (size_t)b * SEQ * HIDDEN + h * HDIM;
  const _Float16* Vtb = Vt + (size_t)(b * NHEADS + h) * HDIM * SEQ;

  // Cache Q fragments: 4 chunks of K-dim (d = 0,32,64,96)
  v16h qf[4];
#pragma unroll
  for (int kc = 0; kc < 4; ++kc) qf[kc] = load_frag(Qb, HIDDEN, kc * 32, lane);

  v8f   od[8] = {};                 // output accumulators, 16 x 128 f32
  float m_row[8], l_row[8];
#pragma unroll
  for (int j = 0; j < 8; ++j) { m_row[j] = -3.402823466e+38f; l_row[j] = 0.0f; }

  const int nlo = lane & 15;
  const int mhi = (lane >> 4) << 3;

  for (int kt = 0; kt < SEQ / 32; ++kt) {
    // ---- scores: S = Q * K_tile^T, two 16x16 tiles over 32 keys ----
    v8f s0 = {}, s1 = {};
    const _Float16* K0 = Kb + (size_t)(kt * 32)      * HIDDEN;
    const _Float16* K1 = Kb + (size_t)(kt * 32 + 16) * HIDDEN;
#pragma unroll
    for (int kc = 0; kc < 4; ++kc) {
      v16h b0 = load_frag(K0, HIDDEN, kc * 32, lane);
      v16h b1 = load_frag(K1, HIDDEN, kc * 32, lane);
      s0 = wmma_f16(qf[kc], b0, s0);
      s1 = wmma_f16(qf[kc], b1, s1);
    }

    // ---- online softmax (rows live across 16-lane halves) ----
    float alpha[8];
#pragma unroll
    for (int j = 0; j < 8; ++j) {
      float v0 = s0[j] * scale;
      float v1 = s1[j] * scale;
      float rm = fmaxf(v0, v1);
#pragma unroll
      for (int off = 1; off < 16; off <<= 1)
        rm = fmaxf(rm, __shfl_xor(rm, off, 32));
      float nm = fmaxf(m_row[j], rm);
      float al = __expf(m_row[j] - nm);
      float p0 = __expf(v0 - nm);
      float p1 = __expf(v1 - nm);
      float rs = p0 + p1;
#pragma unroll
      for (int off = 1; off < 16; off <<= 1)
        rs += __shfl_xor(rs, off, 32);
      l_row[j] = l_row[j] * al + rs;
      m_row[j] = nm;
      alpha[j] = al;
      // store P (C layout) to LDS for A-fragment relayout
      pl[wv][j + mhi][nlo]      = (_Float16)p0;
      pl[wv][j + mhi][nlo + 16] = (_Float16)p1;
    }

    // rescale existing O accumulators (row of od[dt] VGPR j matches alpha[j])
#pragma unroll
    for (int dt = 0; dt < 8; ++dt)
#pragma unroll
      for (int j = 0; j < 8; ++j) od[dt][j] *= alpha[j];

    // reload P as A fragment (16 rows x 32 keys) from LDS: ds_load_b128 x2
    {
      const _Float16* pp = &pl[wv][lane & 15][(lane >> 4) << 3];
      v8h lo = *(const v8h*)pp;
      v8h hi = *(const v8h*)(pp + 16);
      v16h pf;
#pragma unroll
      for (int i = 0; i < 8; ++i) { pf[i] = lo[i]; pf[i + 8] = hi[i]; }

      // ---- PV: O[.,dt*16..] += P(16x32) * V(32x16) ----
#pragma unroll
      for (int dt = 0; dt < 8; ++dt) {
        v16h vb = load_frag(Vtb + (size_t)(dt * 16) * SEQ, SEQ, kt * 32, lane);
        od[dt] = wmma_f16(pf, vb, od[dt]);
      }
    }
  }

  // ---- normalize and write out (B,S,H*D) f16 ----
  _Float16* Ob = O + (size_t)b * SEQ * HIDDEN + (size_t)q0 * HIDDEN + h * HDIM;
#pragma unroll
  for (int j = 0; j < 8; ++j) {
    const float inv = 1.0f / l_row[j];
    const int m = j + mhi;
#pragma unroll
    for (int dt = 0; dt < 8; ++dt)
      Ob[(size_t)m * HIDDEN + dt * 16 + nlo] = (_Float16)(od[dt][j] * inv);
  }
}

// ---------------------------------------------------------------------------
// Host-side orchestration
// ---------------------------------------------------------------------------
extern "C" void kernel_launch(void* const* d_in, const int* in_sizes, int n_in,
                              void* d_out, int out_size, void* d_ws, size_t ws_size,
                              hipStream_t stream) {
  const float* X  = (const float*)d_in[0];
  const float* Wq = (const float*)d_in[1];
  const float* bq = (const float*)d_in[2];
  const float* Wk = (const float*)d_in[3];
  const float* bk = (const float*)d_in[4];
  const float* Wv = (const float*)d_in[5];
  const float* bv = (const float*)d_in[6];
  const float* Wo = (const float*)d_in[7];
  const float* bo = (const float*)d_in[8];
  float* out = (float*)d_out;

  const size_t MB = 1024u * 1024u;
  uint8_t* w = (uint8_t*)d_ws;
  _Float16* Xh  = (_Float16*)(w);             // 16 MB : hidden f16 (4096x2048)
  _Float16* Wqt = (_Float16*)(w + 16 * MB);   //  8 MB : Wq^T f16
  _Float16* Wkt = (_Float16*)(w + 24 * MB);   //  8 MB
  _Float16* Wvt = (_Float16*)(w + 32 * MB);   //  8 MB
  _Float16* Wot = (_Float16*)(w + 40 * MB);   //  8 MB
  _Float16* Qm  = (_Float16*)(w + 48 * MB);   // 16 MB
  _Float16* Km  = (_Float16*)(w + 64 * MB);   // 16 MB
  _Float16* Vm  = (_Float16*)(w + 80 * MB);   // 16 MB
  _Float16* Vtm = (_Float16*)(w + 96 * MB);   // 16 MB : V^T (B,H,D,S)
  _Float16* AO  = (_Float16*)(w + 112 * MB);  // 16 MB : attention output f16

  const int nX = MROWS * HIDDEN;  // 8,388,608

  // 1) casts / transposes
  cast_f32_to_f16<<<(nX + 255) / 256, 256, 0, stream>>>(X, Xh, nX);
  {
    dim3 g(HIDDEN / 32, HIDDEN / 32), blk(32, 8);
    transpose_cast_w<<<g, blk, 0, stream>>>(Wq, Wqt);
    transpose_cast_w<<<g, blk, 0, stream>>>(Wk, Wkt);
    transpose_cast_w<<<g, blk, 0, stream>>>(Wv, Wvt);
    transpose_cast_w<<<g, blk, 0, stream>>>(Wo, Wot);
  }

  // 2) QKV projections (WMMA GEMM, f16 out + bias)
  {
    dim3 g(HIDDEN / 256, MROWS / 64);
    gemm_bias_kernel<_Float16><<<g, 256, 0, stream>>>(Xh, Wqt, bq, Qm, MROWS, HIDDEN, HIDDEN);
    gemm_bias_kernel<_Float16><<<g, 256, 0, stream>>>(Xh, Wkt, bk, Km, MROWS, HIDDEN, HIDDEN);
    gemm_bias_kernel<_Float16><<<g, 256, 0, stream>>>(Xh, Wvt, bv, Vm, MROWS, HIDDEN, HIDDEN);
  }

  // 3) V transpose for contiguous PV B-fragments
  transpose_v<<<(nX + 255) / 256, 256, 0, stream>>>(Vm, Vtm);

  // 4) attention (flash-style, 4 waves/block, one wave per 16 query rows)
  {
    dim3 g(SEQ / 64, NHEADS, BATCH);
    attn_kernel<<<g, 128, 0, stream>>>(Qm, Km, Vtm, AO);
  }

  // 5) output projection (f32 out + bias)
  {
    dim3 g(HIDDEN / 256, MROWS / 64);
    gemm_bias_kernel<float><<<g, 256, 0, stream>>>(AO, Wot, bo, out, MROWS, HIDDEN, HIDDEN);
  }
}
